// MultiHeadSelfAttention_77146202571298
// MI455X (gfx1250) — compile-verified, matrix operand order checked by reading, not run on hardware
//
#include <hip/hip_runtime.h>

typedef __attribute__((ext_vector_type(16))) _Float16 v16h;
typedef __attribute__((ext_vector_type(8)))  float    v8f;

constexpr int DM     = 768;     // d_model
constexpr int NTOK   = 4 * 2048;
constexpr int SEQ    = 2048;
constexpr int NH     = 12;
constexpr int DH     = 64;
constexpr int KT     = 32;          // WMMA K per step (f16)
constexpr int KSTEPS = DM / KT;     // 24
constexpr int KBLK   = SEQ / 32;    // 64 key blocks

// batched group-of-32 xor swizzle max-reduction step over 8 independent rows:
// 8 ds_swizzles issue back-to-back, one wait, 8 maxes.
template <int XM>
__device__ inline void red_max8(float (&rm)[8]) {
    float t[8];
#pragma unroll
    for (int j = 0; j < 8; ++j)
        t[j] = __int_as_float(
            __builtin_amdgcn_ds_swizzle(__float_as_int(rm[j]), (XM << 10) | 0x1f));
#pragma unroll
    for (int j = 0; j < 8; ++j) rm[j] = fmaxf(rm[j], t[j]);
}

// ---------------------------------------------------------------------------
// fp32 -> f16 elementwise convert (x)
// ---------------------------------------------------------------------------
__global__ void cvt_f32_f16(const float* __restrict__ in, _Float16* __restrict__ out, int n) {
    int i = (blockIdx.x * blockDim.x + threadIdx.x) * 4;
    if (i < n) {
        float4 v = *reinterpret_cast<const float4*>(in + i);
        out[i + 0] = (_Float16)v.x;
        out[i + 1] = (_Float16)v.y;
        out[i + 2] = (_Float16)v.z;
        out[i + 3] = (_Float16)v.w;
    }
}

// ---------------------------------------------------------------------------
// Pack W [out=768][in=768] (row-major fp32) into per-lane WMMA B-fragments.
// ---------------------------------------------------------------------------
__global__ void pack_w_bfrag(const float* __restrict__ W, _Float16* __restrict__ Wp) {
    const int t    = blockIdx.x * blockDim.x + threadIdx.x;   // 36864 threads
    const int lane = t & 31;
    const int tile = t >> 5;                                  // 0..1151
    const int nt   = tile % (DM / 16);
    const int kt   = tile / (DM / 16);
    const int n    = nt * 16 + (lane & 15);
    const int k0   = kt * 32 + (lane >> 4) * 16;
    const float* src = &W[(size_t)n * DM + k0];
    _Float16*    dst = &Wp[((size_t)tile * 32 + lane) * 16];
#pragma unroll
    for (int i = 0; i < 16; ++i) dst[i] = (_Float16)src[i];
}

// ---------------------------------------------------------------------------
// GEMM: Y[8192,768] = Xf16 * W^T + bias. WG tile 64x128, 8 waves, 16x64 strip
// per wave (4 WMMA accs, 24 K-steps). Double-buffered LDS X tile, one barrier
// per K-step, next tile's global load in flight under the WMMAs.
// ---------------------------------------------------------------------------
template <bool F32OUT>
__global__ void __launch_bounds__(256)
gemm_bias_768(const _Float16* __restrict__ X,
              const _Float16* __restrict__ Wp,
              const float*    __restrict__ bias,
              float*          __restrict__ Yf32,
              _Float16*       __restrict__ Yf16) {
    __shared__ __align__(32) _Float16 Xs[2][64 * KT];   // 2 x 4 KB

    const int lane  = threadIdx.x & 31;
    const int wave  = threadIdx.x >> 5;
    const int mw    = wave & 3;
    const int nw    = wave >> 2;
    const int m0    = blockIdx.y * 64;
    const int n0    = blockIdx.x * 128;
    const int lrow  = lane & 15;
    const int lhalf = lane >> 4;

    const int sidx = threadIdx.x * 8;   // staging: 256 threads x 8 halfs
    const int sr   = sidx >> 5;
    const int sc   = sidx & 31;

    uint4 xreg;
    auto gload = [&](int kt) {
        xreg = *reinterpret_cast<const uint4*>(&X[(size_t)(m0 + sr) * DM + kt * KT + sc]);
    };
    auto sstore = [&](int buf) {
        *reinterpret_cast<uint4*>(&Xs[buf][sr * KT + sc]) = xreg;
    };

    v8f c[4] = {};

    gload(0);
    sstore(0);
    for (int kt = 0; kt < KSTEPS; ++kt) {
        __syncthreads();
        const int cur = kt & 1;
        if (kt + 1 < KSTEPS) {
            gload(kt + 1);
            if (kt + 2 < KSTEPS)
                __builtin_prefetch(&X[(size_t)(m0 + sr) * DM + (kt + 2) * KT + sc], 0, 0);
        }

        const v16h a = *reinterpret_cast<const v16h*>(
            &Xs[cur][(mw * 16 + lrow) * KT + lhalf * 16]);
#pragma unroll
        for (int j = 0; j < 4; ++j) {
            const int ntG = (n0 >> 4) + nw * 4 + j;
            const v16h b  = *reinterpret_cast<const v16h*>(
                &Wp[((size_t)(kt * (DM / 16) + ntG) * 32 + lane) * 16]);
            c[j] = __builtin_amdgcn_wmma_f32_16x16x32_f16(
                false, a, false, b, (short)0, c[j], false, false);
        }

        if (kt + 1 < KSTEPS) sstore(cur ^ 1);
    }

#pragma unroll
    for (int j = 0; j < 4; ++j) {
        const int col  = n0 + nw * 64 + j * 16 + lrow;
        const float bv = bias[col];
#pragma unroll
        for (int jv = 0; jv < 8; ++jv) {
            const int row   = m0 + mw * 16 + jv + 8 * lhalf;
            const float val = c[j][jv] + bv;
            if constexpr (F32OUT) Yf32[(size_t)row * DM + col] = val;
            else                  Yf16[(size_t)row * DM + col] = (_Float16)val;
        }
    }
}

// ---------------------------------------------------------------------------
// Flash attention. Grid (16, 12, 4), 8 waves/block, one 16-query strip per
// wave. Double-buffered shared K/V 32-key tiles, one barrier per block.
// Per block: 4 score WMMAs + 4 PV WMMAs + 1 rowsum WMMA (P x ones).
// ---------------------------------------------------------------------------
__global__ void __launch_bounds__(256, 1)
attn_flash(const _Float16* __restrict__ Q,
           const _Float16* __restrict__ K,
           const _Float16* __restrict__ V,
           const int*      __restrict__ mask,
           const float*    __restrict__ temp,
           _Float16*       __restrict__ O) {
    __shared__ __align__(32) _Float16 Ks [2][32 * DH];    // [key][d]
    __shared__ __align__(32) _Float16 Vts[2][DH * 32];    // [d][key] transposed
    __shared__ __align__(32) _Float16 Ps [8][16 * 32];    // per-wave P scratch
    __shared__ float Mb[SEQ];                             // additive mask bias

    const int lane  = threadIdx.x & 31;
    const int wave  = threadIdx.x >> 5;
    const int b     = blockIdx.z;
    const int h     = blockIdx.y;
    const int q0    = (blockIdx.x * 8 + wave) * 16;
    const int lrow  = lane & 15;
    const int lhalf = lane >> 4;
    const float scale = temp[h] * 0.125f;     // temperature / sqrt(64)

    const size_t tokBase = (size_t)b * SEQ;

    // stage additive mask bias once (visible after first in-loop barrier)
    {
        const int base = threadIdx.x * 8;
#pragma unroll
        for (int i = 0; i < 8; ++i)
            Mb[base + i] = mask[b * SEQ + base + i] ? 0.0f : -1.0e9f * scale;
    }

    // Q fragments pre-scaled by temperature/sqrt(d)
    const _Float16 hs = (_Float16)scale;
    v16h aq[2];
#pragma unroll
    for (int f = 0; f < 2; ++f) {
        aq[f] = *reinterpret_cast<const v16h*>(
            &Q[(tokBase + q0 + lrow) * DM + h * DH + f * 32 + lhalf * 16]);
        aq[f] = aq[f] * hs;
    }

    v16h ones;
#pragma unroll
    for (int i = 0; i < 16; ++i) ones[i] = (_Float16)1.0f;

    v8f   acc[4] = {};
    v8f   accS   = {};
    float mrun[8];
#pragma unroll
    for (int jv = 0; jv < 8; ++jv) mrun[jv] = -3.0e38f;

    const int skey = threadIdx.x >> 3;   // K/V staging: 8 threads per key
    const int sch  = threadIdx.x & 7;

    uint4 kreg, vreg;
    auto gload = [&](int kb) {
        const size_t g = (tokBase + kb * 32 + skey) * DM + h * DH + sch * 8;
        kreg = *reinterpret_cast<const uint4*>(&K[g]);
        vreg = *reinterpret_cast<const uint4*>(&V[g]);
    };
    auto sstore = [&](int buf) {
        *reinterpret_cast<uint4*>(&Ks[buf][skey * DH + sch * 8]) = kreg;
        const _Float16* vp = reinterpret_cast<const _Float16*>(&vreg);
#pragma unroll
        for (int i = 0; i < 8; ++i) Vts[buf][(sch * 8 + i) * 32 + skey] = vp[i];
    };

    gload(0);
    sstore(0);
    for (int kb = 0; kb < KBLK; ++kb) {
        __syncthreads();
        const int cur = kb & 1;
        if (kb + 1 < KBLK) gload(kb + 1);     // in flight under the WMMAs

        // scores (pre-scaled), two 16-key column sub-tiles
        v8f sc0 = {}, sc1 = {};
#pragma unroll
        for (int f = 0; f < 2; ++f) {
            const v16h bk0 = *reinterpret_cast<const v16h*>(
                &Ks[cur][lrow * DH + f * 32 + lhalf * 16]);
            const v16h bk1 = *reinterpret_cast<const v16h*>(
                &Ks[cur][(16 + lrow) * DH + f * 32 + lhalf * 16]);
            sc0 = __builtin_amdgcn_wmma_f32_16x16x32_f16(
                false, aq[f], false, bk0, (short)0, sc0, false, false);
            sc1 = __builtin_amdgcn_wmma_f32_16x16x32_f16(
                false, aq[f], false, bk1, (short)0, sc1, false, false);
        }

        const int kk0 = kb * 32;
        const float bias0 = Mb[kk0 + lrow];
        const float bias1 = Mb[kk0 + 16 + lrow];

        // batched row-max butterflies (8 swizzles per step, one wait each)
        float rm[8];
#pragma unroll
        for (int jv = 0; jv < 8; ++jv) {
            sc0[jv] += bias0;
            sc1[jv] += bias1;
            rm[jv] = fmaxf(sc0[jv], sc1[jv]);
        }
        red_max8<8>(rm);
        red_max8<4>(rm);
        red_max8<2>(rm);
        red_max8<1>(rm);

#pragma unroll
        for (int jv = 0; jv < 8; ++jv) {
            const float mnew = fmaxf(mrun[jv], rm[jv]);
            const float cr   = __expf(mrun[jv] - mnew);
            mrun[jv] = mnew;
            const float p0 = __expf(sc0[jv] - mnew);
            const float p1 = __expf(sc1[jv] - mnew);
            const int prow = jv + 8 * lhalf;              // C-layout row
            Ps[wave][prow * 32 + lrow]      = (_Float16)p0;
            Ps[wave][prow * 32 + 16 + lrow] = (_Float16)p1;
            accS[jv] *= cr;
#pragma unroll
            for (int nt = 0; nt < 4; ++nt) acc[nt][jv] *= cr;
        }

        // P as A fragment; V^T + ones as B fragments
        const v16h ap = *reinterpret_cast<const v16h*>(&Ps[wave][lrow * 32 + lhalf * 16]);
        accS = __builtin_amdgcn_wmma_f32_16x16x32_f16(
            false, ap, false, ones, (short)0, accS, false, false);
#pragma unroll
        for (int nt = 0; nt < 4; ++nt) {
            const v16h bv = *reinterpret_cast<const v16h*>(
                &Vts[cur][(nt * 16 + lrow) * 32 + lhalf * 16]);
            acc[nt] = __builtin_amdgcn_wmma_f32_16x16x32_f16(
                false, ap, false, bv, (short)0, acc[nt], false, false);
        }

        if (kb + 1 < KBLK) sstore(cur ^ 1);   // waits loadcnt here, post-compute
    }

    // normalize + store f16 ([token][768], heads interleaved)
#pragma unroll
    for (int jv = 0; jv < 8; ++jv) {
        const float inv = 1.0f / accS[jv];
        const int   row = q0 + jv + 8 * lhalf;
#pragma unroll
        for (int nt = 0; nt < 4; ++nt) {
            const float ov = acc[nt][jv] * inv;
            O[(tokBase + row) * DM + h * DH + nt * 16 + lrow] = (_Float16)ov;
        }
    }
}

// ---------------------------------------------------------------------------
extern "C" void kernel_launch(void* const* d_in, const int* in_sizes, int n_in,
                              void* d_out, int out_size, void* d_ws, size_t ws_size,
                              hipStream_t stream) {
    const float* x    = (const float*)d_in[0];
    const int*   mask = (const int*)  d_in[1];
    const float* Wq   = (const float*)d_in[2];
    const float* bq   = (const float*)d_in[3];
    const float* Wk   = (const float*)d_in[4];
    const float* bk   = (const float*)d_in[5];
    const float* Wv   = (const float*)d_in[6];
    const float* bv   = (const float*)d_in[7];
    const float* Wo   = (const float*)d_in[8];
    const float* bo   = (const float*)d_in[9];
    const float* temp = (const float*)d_in[10];
    float* out = (float*)d_out;

    size_t off = 0;
    auto walloc = [&](size_t elems) {
        void* p = (char*)d_ws + off;
        off = (off + elems * sizeof(_Float16) + 255) & ~(size_t)255;
        return (_Float16*)p;
    };
    _Float16* Xf  = walloc((size_t)NTOK * DM);
    _Float16* Wqp = walloc((size_t)DM * DM);
    _Float16* Wkp = walloc((size_t)DM * DM);
    _Float16* Wvp = walloc((size_t)DM * DM);
    _Float16* Wop = walloc((size_t)DM * DM);
    _Float16* Qf  = walloc((size_t)NTOK * DM);
    _Float16* Kf  = walloc((size_t)NTOK * DM);
    _Float16* Vf  = walloc((size_t)NTOK * DM);
    _Float16* Af  = Xf;   // attention output reuses x_f16 (dead after QKV GEMMs)

    const int nelem = NTOK * DM;
    cvt_f32_f16<<<nelem / (256 * 4), 256, 0, stream>>>(x, Xf, nelem);

    const int packThreads = (DM / 32) * (DM / 16) * 32;   // 36864
    pack_w_bfrag<<<packThreads / 256, 256, 0, stream>>>(Wq, Wqp);
    pack_w_bfrag<<<packThreads / 256, 256, 0, stream>>>(Wk, Wkp);
    pack_w_bfrag<<<packThreads / 256, 256, 0, stream>>>(Wv, Wvp);
    pack_w_bfrag<<<packThreads / 256, 256, 0, stream>>>(Wo, Wop);

    dim3 ggrid(DM / 128, NTOK / 64);   // (6, 128)
    gemm_bias_768<false><<<ggrid, 256, 0, stream>>>(Xf, Wqp, bq, nullptr, Qf);
    gemm_bias_768<false><<<ggrid, 256, 0, stream>>>(Xf, Wkp, bk, nullptr, Kf);
    gemm_bias_768<false><<<ggrid, 256, 0, stream>>>(Xf, Wvp, bv, nullptr, Vf);

    attn_flash<<<dim3(SEQ / (16 * 8), NH, 4), 256, 0, stream>>>(Qf, Kf, Vf, mask, temp, Af);

    gemm_bias_768<true><<<ggrid, 256, 0, stream>>>(Af, Wop, bo, out, nullptr);
}
